// attention_layer_36687610642963
// MI455X (gfx1250) — matrix operand order repaired and checked by reading, not compiled, not run on hardware
//
#include <hip/hip_runtime.h>

// CDNA5 / gfx1250 WMMA types (wave32)
typedef __attribute__((ext_vector_type(16))) __bf16 v16bf;
typedef __attribute__((ext_vector_type(8)))  float  v8f;

#define B_      32
#define N_      1024
#define F_      128
#define NEG_INF (-9.0e15f)
#define ALPHA_  0.2f

// Issue 4x GLOBAL_LOAD_ASYNC_TO_LDS_B128 copying 64 contiguous bytes
// global -> LDS. Per ISA 08_async_tensor 4.4, INST_OFFSET is added to BOTH
// the LDS dest (VDST vgpr) and the global src (VADDR vgpr pair).
// Tracked by ASYNCcnt; in-order among async loads.
__device__ __forceinline__ void async_copy64(const void* lds_dst,
                                             const void* gsrc) {
  // Generic pointers into the LDS aperture carry the wave-relative LDS byte
  // offset in their low 32 bits (ISA 10.2).
  const unsigned lofs = (unsigned)(unsigned long long)lds_dst;
  const unsigned long long ga = (unsigned long long)gsrc;
  asm volatile(
      "global_load_async_to_lds_b128 %0, %1, off\n\t"
      "global_load_async_to_lds_b128 %0, %1, off offset:16\n\t"
      "global_load_async_to_lds_b128 %0, %1, off offset:32\n\t"
      "global_load_async_to_lds_b128 %0, %1, off offset:48"
      :: "v"(lofs), "v"(ga) : "memory");
}

// -------------------------------------------------------------------------
// Kernel 1: h = X @ W^T  (bf16 WMMA, f32 accum), plus s = h.a_src, t = h.a_dst
// Grid: (B*N/128) blocks of 256 threads (8 waves). Each wave owns a 16-row
// stripe of the 128x128 output tile (8 accumulator tiles along d).
// -------------------------------------------------------------------------
__global__ __launch_bounds__(256) void gat_h_kernel(
    const float* __restrict__ Xg,   // (B*N, F) f32
    const float* __restrict__ Wg,   // (F, F)   f32, row-major (d, f)
    const float* __restrict__ Ag,   // (2F,)    f32: a_src | a_dst
    __bf16*      __restrict__ hb,   // (B*N, F) bf16 out
    float*       __restrict__ Sg,   // (B*N,)   f32 out
    float*       __restrict__ Tg)   // (B*N,)   f32 out
{
  __shared__ __align__(32) __bf16 Wl[F_ * F_];   // 32 KB, row-major (d, f)

  const int tid    = threadIdx.x;
  const int wave   = tid >> 5;
  const int lane   = tid & 31;
  const int lane16 = lane & 15;
  const int kh     = lane >> 4;          // which K-half this lane holds
  const int g0     = blockIdx.x * 128;   // flattened (b*N + n) row base

  // Stage W into LDS as bf16 (coalesced float4 reads)
  for (int i = tid; i < (F_ * F_) / 4; i += 256) {
    const float4 wv = reinterpret_cast<const float4*>(Wg)[i];
    const int o = i * 4;
    Wl[o + 0] = (__bf16)wv.x;  Wl[o + 1] = (__bf16)wv.y;
    Wl[o + 2] = (__bf16)wv.z;  Wl[o + 3] = (__bf16)wv.w;
  }
  __syncthreads();

  const int arow = g0 + wave * 16 + lane16;   // A-matrix row for this lane

  v8f acc[8] = {};   // 8 d-tiles of 16x16 f32

  #pragma unroll
  for (int f0 = 0; f0 < F_; f0 += 32) {
    // A fragment: 16x32 bf16.  Lane holds K {kh*16+2i(+1)} (contiguous,
    // matching B).  Convert f32 -> bf16 on the fly (float2 loads).
    v16bf afrag;
    #pragma unroll
    for (int i = 0; i < 8; ++i) {
      const int k0 = f0 + kh * 16 + 2 * i;
      const float2 xv =
          *reinterpret_cast<const float2*>(&Xg[(size_t)arow * F_ + k0]);
      afrag[2 * i]     = (__bf16)xv.x;
      afrag[2 * i + 1] = (__bf16)xv.y;
    }
    #pragma unroll
    for (int dt = 0; dt < 8; ++dt) {
      // B fragment: B[k][n=d] = W[d][f0+k]; lane's 16 K-values are the 16
      // contiguous bf16 at Wl[d*F + f0 + kh*16]  (32 B aligned ds load).
      const int d = dt * 16 + lane16;
      const v16bf bfrag =
          *reinterpret_cast<const v16bf*>(&Wl[d * F_ + f0 + kh * 16]);
      acc[dt] = __builtin_amdgcn_wmma_f32_16x16x32_bf16(
          false, afrag, false, bfrag, (short)0, acc[dt], false, false);
    }
  }

  // Epilogue: store h as bf16; fold in s/t dot products from f32 accumulators.
  float as[8], ad[8];
  #pragma unroll
  for (int dt = 0; dt < 8; ++dt) {
    as[dt] = Ag[dt * 16 + lane16];
    ad[dt] = Ag[F_ + dt * 16 + lane16];
  }
  float sp[8], tp[8];
  #pragma unroll
  for (int r = 0; r < 8; ++r) { sp[r] = 0.f; tp[r] = 0.f; }

  #pragma unroll
  for (int dt = 0; dt < 8; ++dt) {
    const int d = dt * 16 + lane16;
    #pragma unroll
    for (int r = 0; r < 8; ++r) {
      // C/D layout: VGPR r -> row (r + 8*kh), col lane16
      const float hv   = acc[dt][r];
      const int   grow = g0 + wave * 16 + r + 8 * kh;
      hb[(size_t)grow * F_ + d] = (__bf16)hv;
      sp[r] += hv * as[dt];
      tp[r] += hv * ad[dt];
    }
  }
  // Reduce across the 16 lanes that share each row (xor masks stay in-half)
  #pragma unroll
  for (int m = 1; m <= 8; m <<= 1) {
    #pragma unroll
    for (int r = 0; r < 8; ++r) {
      sp[r] += __shfl_xor(sp[r], m, 32);
      tp[r] += __shfl_xor(tp[r], m, 32);
    }
  }
  if (lane16 == 0) {
    #pragma unroll
    for (int r = 0; r < 8; ++r) {
      const int grow = g0 + wave * 16 + r + 8 * kh;
      Sg[grow] = sp[r];
      Tg[grow] = tp[r];
    }
  }
}

// -------------------------------------------------------------------------
// Kernel 2: flash-style masked softmax-attention, out = softmax(mask) @ h.
// Grid: (N/128, B) blocks of 256 threads. Single pass over j (adj read once
// from HBM, streamed via double-buffered GLOBAL_LOAD_ASYNC_TO_LDS_B128),
// online max/sum, bf16 WMMA for attn @ h with f32 accumulation.
// -------------------------------------------------------------------------
__global__ __launch_bounds__(256) void gat_attn_kernel(
    const float*  __restrict__ adj,  // (B, N, N) f32
    const __bf16* __restrict__ hb,   // (B*N, F)  bf16
    const float*  __restrict__ Sg,   // (B*N,)
    const float*  __restrict__ Tg,   // (B*N,)
    float*        __restrict__ out)  // (B, N, F) f32
{
  __shared__ float t_l[N_];                          //  4 KB
  __shared__ __align__(32) float  adjR[2][128 * 32]; // 32 KB, double buffer
  __shared__ __align__(32) __bf16 hT[128 * 32];      //  8 KB, transposed [d][k]

  const int tid    = threadIdx.x;
  const int wave   = tid >> 5;
  const int lane   = tid & 31;
  const int lane16 = lane & 15;
  const int kh     = lane >> 4;
  const int b      = blockIdx.y;
  const int i0     = blockIdx.x * 128;
  const int NT     = N_ / 32;          // 32 j-tiles

  for (int i = tid; i < N_; i += 256) t_l[i] = Tg[b * N_ + i];
  const float srow = Sg[b * N_ + i0 + wave * 16 + lane16];

  const float* adjB = adj + (size_t)b * N_ * N_ + (size_t)i0 * N_;

  // Per-thread staging coordinates
  const int sr = tid >> 1;             // adj row within block   (0..127)
  const int sc = (tid & 1) * 16;       // adj col half           (0 or 16)
  const int hk = tid >> 3;             // h tile row k           (0..31)
  const int hd = (tid & 7) * 16;       // h tile col base d      (0..112)

  // ---- prologue: async adj tile 0 -> buf 0 ; h tile 0 -> regs ----
  async_copy64(&adjR[0][sr * 32 + sc], adjB + (size_t)sr * N_ + sc);
  uint4 h0, h1;
  {
    const uint4* src = reinterpret_cast<const uint4*>(
        hb + (size_t)(b * N_ + hk) * F_ + hd);
    h0 = src[0]; h1 = src[1];
  }

  float M = NEG_INF;   // running max for row (wave*16 + lane16)
  float L = 0.f;       // running denom for same row
  v8f acc[8] = {};     // 16x128 f32 stripe

  for (int jt = 0; jt < NT; ++jt) {
    const int cur = jt & 1;
    const int nxt = cur ^ 1;
    const int j0  = jt * 32;

    // ---- stage h tile (regs, fetched last iter) transposed into hT[d][k] ----
    {
      __bf16 tmp[16];
      *reinterpret_cast<uint4*>(&tmp[0]) = h0;
      *reinterpret_cast<uint4*>(&tmp[8]) = h1;
      #pragma unroll
      for (int q = 0; q < 16; ++q) hT[(hd + q) * 32 + hk] = tmp[q];
    }

    // ---- prefetch next tile: adj via async->LDS, h via regs ----
    if (jt + 1 < NT) {
      async_copy64(&adjR[nxt][sr * 32 + sc],
                   adjB + (size_t)sr * N_ + (j0 + 32) + sc);
      const uint4* src = reinterpret_cast<const uint4*>(
          hb + (size_t)(b * N_ + j0 + 32 + hk) * F_ + hd);
      h0 = src[0]; h1 = src[1];
      // <=4 outstanding allowed: the 4 just issued. Older (current buffer)
      // asyncs complete in order -> current tile's LDS data is resident.
      asm volatile("s_wait_asynccnt 0x4" ::: "memory");
    } else {
      asm volatile("s_wait_asynccnt 0x0" ::: "memory");
    }
    __syncthreads();   // hT stores + every wave's asyncs for `cur` visible

    const float* aR = adjR[cur];

    // ---- masked leaky-relu scores, directly in A-fragment lane layout ----
    const int myrow = wave * 16 + lane16;
    float lr[16];
    int   amask = 0;
    float tmax  = NEG_INF;
    #pragma unroll
    for (int e = 0; e < 16; ++e) {
      const int ke = kh * 16 + e;
      float x = srow + t_l[j0 + ke];
      x = (x > 0.f) ? x : ALPHA_ * x;
      lr[e] = x;
      const bool on = (aR[myrow * 32 + ke] != 0.f);
      amask |= ((int)on) << e;
      tmax = fmaxf(tmax, on ? x : NEG_INF);
    }
    tmax = fmaxf(tmax, __shfl_xor(tmax, 16, 32));   // both K-halves of row
    const float Mnew  = fmaxf(M, tmax);
    const float scale = __expf(M - Mnew);           // ==1 if unchanged
    M = Mnew;

    v16bf afrag;
    float tsum = 0.f;
    #pragma unroll
    for (int e = 0; e < 16; ++e) {
      const float p = ((amask >> e) & 1) ? __expf(lr[e] - M) : 0.f;
      tsum += p;
      afrag[e] = (__bf16)p;
    }
    tsum += __shfl_xor(tsum, 16, 32);
    L = L * scale + tsum;

    // rescale accumulators: row r+8*kh's factor lives in lane (r+8*kh)
    #pragma unroll
    for (int r = 0; r < 8; ++r) {
      const float scr = __shfl(scale, r + 8 * kh, 32);
      #pragma unroll
      for (int dt = 0; dt < 8; ++dt) acc[dt][r] *= scr;
    }

    // ---- attn-tile @ h-tile : 8 WMMAs, B-fragments contiguous in hT ----
    #pragma unroll
    for (int dt = 0; dt < 8; ++dt) {
      const int d = dt * 16 + lane16;
      const v16bf bfrag =
          *reinterpret_cast<const v16bf*>(&hT[d * 32 + kh * 16]);
      acc[dt] = __builtin_amdgcn_wmma_f32_16x16x32_bf16(
          false, afrag, false, bfrag, (short)0, acc[dt], false, false);
    }

    // All waves done reading `cur` buffers before next iteration overwrites
    // adjR[cur] (async) and hT (ds stores).
    __syncthreads();
  }

  // ---- normalize by softmax denominator and store ----
  float linv[8];
  #pragma unroll
  for (int r = 0; r < 8; ++r) {
    const float Lr = __shfl(L, r + 8 * kh, 32);  // self-loop => Lr > 0
    linv[r] = 1.0f / Lr;
  }
  #pragma unroll
  for (int dt = 0; dt < 8; ++dt) {
    #pragma unroll
    for (int r = 0; r < 8; ++r) {
      out[(size_t)(b * N_ + i0 + wave * 16 + r + 8 * kh) * F_ +
          dt * 16 + lane16] = acc[dt][r] * linv[r];
    }
  }
}

// -------------------------------------------------------------------------
extern "C" void kernel_launch(void* const* d_in, const int* in_sizes, int n_in,
                              void* d_out, int out_size, void* d_ws,
                              size_t ws_size, hipStream_t stream) {
  (void)in_sizes; (void)n_in; (void)out_size; (void)ws_size;
  const float* adj = (const float*)d_in[0];   // (B, N, N)
  const float* X   = (const float*)d_in[1];   // (B, N, F)
  const float* W   = (const float*)d_in[2];   // (F, F)
  const float* a   = (const float*)d_in[3];   // (2F, 1)
  float* out = (float*)d_out;

  // workspace: hb (bf16, 8 MB) | s (128 KB) | t (128 KB)
  __bf16* hb = (__bf16*)d_ws;
  float*  Sg = (float*)((char*)d_ws + (size_t)B_ * N_ * F_ * sizeof(__bf16));
  float*  Tg = Sg + B_ * N_;

  gat_h_kernel<<<dim3((B_ * N_) / 128), dim3(256), 0, stream>>>(
      X, W, a, hb, Sg, Tg);
  gat_attn_kernel<<<dim3(N_ / 128, B_), dim3(256), 0, stream>>>(
      adj, hb, Sg, Tg, out);
}
